// CTDAutoEncoder_22771916604023
// MI455X (gfx1250) — matrix operand order repaired and checked by reading, not compiled, not run on hardware
//
#include <hip/hip_runtime.h>
#include <hip/hip_bf16.h>

typedef __attribute__((ext_vector_type(16))) __bf16 v16bf;
typedef __attribute__((ext_vector_type(8)))  float  v8f;

#define C_DIM 128   // feature dim
#define K_CB  1024  // codebook entries

// ---------------------------------------------------------------------------
// Kernel 1: codebook fp32 -> bf16 hi/lo split + ||c||^2 per row
// grid = K_CB blocks, 128 threads
// ---------------------------------------------------------------------------
__global__ __launch_bounds__(128) void vq_prep_kernel(
    const float* __restrict__ cb, __bf16* __restrict__ cbhi,
    __bf16* __restrict__ cblo, float* __restrict__ cbsq) {
  const int row = blockIdx.x;
  const int t   = threadIdx.x;
  const float c = cb[row * C_DIM + t];
  const __bf16 h = (__bf16)c;
  const __bf16 l = (__bf16)(c - (float)h);
  cbhi[row * C_DIM + t] = h;
  cblo[row * C_DIM + t] = l;

  __shared__ float sm[C_DIM];
  sm[t] = c * c;
  __syncthreads();
  #pragma unroll
  for (int s = 64; s > 0; s >>= 1) {
    if (t < s) sm[t] += sm[t + s];
    __syncthreads();
  }
  if (t == 0) cbsq[row] = sm[0];
}

// ---------------------------------------------------------------------------
// Kernel 2: per-wave 16-row tile; WMMA bf16x3 GEMM vs all 1024 codes,
// fused argmin/min + code gather. Three independent accumulator chains
// (hi*hi, hi*lo, lo*hi) to break the WMMA->WMMA serial dependence.
// 256 threads = 8 waves -> 128 rows per block.
// ---------------------------------------------------------------------------
__global__ __launch_bounds__(256) void vq_search_kernel(
    const float* __restrict__ x, const float* __restrict__ cb,
    const __bf16* __restrict__ cbhi, const __bf16* __restrict__ cblo,
    const float* __restrict__ cbsq,
    float* __restrict__ codes, int* __restrict__ flat_idx,
    int* __restrict__ idx_out, float* __restrict__ dists) {
  const int lane   = threadIdx.x & 31;
  const int wave   = threadIdx.x >> 5;
  const long Rbase = ((long)blockIdx.x * 8 + wave) * 16;

  const int r      = lane & 15;        // row-in-tile (A) / column (B,C,D)
  const int grpHi  = lane >> 4;        // 0: lanes 0-15, 1: lanes 16-31
  const int ka     = grpHi * 16;       // A-fragment K base within 32-chunk
  const int kb     = grpHi * 16;       // B-fragment K base within 32-chunk

  // ---- Load & convert the 16x128 A tile into bf16 hi/lo fragments ----
  v16bf a_hi[4], a_lo[4];
  float partial = 0.f;
  const float* xr = x + (Rbase + r) * C_DIM;
  #pragma unroll
  for (int ks = 0; ks < 4; ++ks) {
    const int kk = ks * 32 + ka;
    float4 f0 = *(const float4*)(xr + kk + 0);
    float4 f1 = *(const float4*)(xr + kk + 4);
    float4 f2 = *(const float4*)(xr + kk + 8);
    float4 f3 = *(const float4*)(xr + kk + 12);
    float vals[16] = {f0.x, f0.y, f0.z, f0.w, f1.x, f1.y, f1.z, f1.w,
                      f2.x, f2.y, f2.z, f2.w, f3.x, f3.y, f3.z, f3.w};
    #pragma unroll
    for (int e = 0; e < 16; ++e) {
      const float fv = vals[e];
      const __bf16 h = (__bf16)fv;
      a_hi[ks][e] = h;
      a_lo[ks][e] = (__bf16)(fv - (float)h);
      partial = fmaf(fv, fv, partial);
    }
  }
  // ||x_row||^2: lane l + lane l^16 together cover all 128 k's of row r.
  const float rowsq = partial + __shfl_xor(partial, 16, 32);

  // ---- Sweep all 64 codebook column tiles ----
  float bval[8];
  int   bidx[8];
  #pragma unroll
  for (int v = 0; v < 8; ++v) { bval[v] = 3.402823466e+38f; bidx[v] = 0; }

  for (int t = 0; t < K_CB / 16; ++t) {
    const int j = t * 16 + r;  // codebook row this lane-column evaluates
    v8f acc_hh = {};           // hi(A) * hi(B)
    v8f acc_hl = {};           // hi(A) * lo(B)
    v8f acc_lh = {};           // lo(A) * hi(B)
    #pragma unroll
    for (int ks = 0; ks < 4; ++ks) {
      const __bf16* ph = cbhi + (size_t)j * C_DIM + ks * 32 + kb;
      const __bf16* pl = cblo + (size_t)j * C_DIM + ks * 32 + kb;
      v16bf bh = *(const v16bf*)ph;
      v16bf bl = *(const v16bf*)pl;
      acc_hh = __builtin_amdgcn_wmma_f32_16x16x32_bf16(
          false, a_hi[ks], false, bh, (short)0, acc_hh, false, false);
      acc_hl = __builtin_amdgcn_wmma_f32_16x16x32_bf16(
          false, a_hi[ks], false, bl, (short)0, acc_hl, false, false);
      acc_lh = __builtin_amdgcn_wmma_f32_16x16x32_bf16(
          false, a_lo[ks], false, bh, (short)0, acc_lh, false, false);
    }
    const float cq = cbsq[j];
    #pragma unroll
    for (int v = 0; v < 8; ++v) {
      // s = cq - 2*(hh + hl + lh), folded into a 3-FMA chain
      float s = fmaf(-2.f, acc_lh[v], cq);
      s = fmaf(-2.f, acc_hl[v], s);
      s = fmaf(-2.f, acc_hh[v], s);
      if (s < bval[v]) { bval[v] = s; bidx[v] = j; }  // strict <: first-min wins
    }
  }

  // ---- Reduce across the 16 lane-columns of each half (rows v / v+8) ----
  #pragma unroll
  for (int v = 0; v < 8; ++v) {
    #pragma unroll
    for (int m = 1; m <= 8; m <<= 1) {
      const float ov = __shfl_xor(bval[v], m, 32);
      const int   oi = __shfl_xor(bidx[v], m, 32);
      if (ov < bval[v] || (ov == bval[v] && oi < bidx[v])) {
        bval[v] = ov; bidx[v] = oi;
      }
    }
  }

  // ---- Scalar outputs: flat_idx, idx, distances ----
  #pragma unroll
  for (int v = 0; v < 8; ++v) {
    const float sq_lo = __shfl(rowsq, v, 32);      // row v
    const float sq_hi = __shfl(rowsq, v + 8, 32);  // row v+8
    if (lane == 0) {
      const long row = Rbase + v;
      flat_idx[row] = bidx[v];
      idx_out[row]  = bidx[v];
      dists[row]    = bval[v] + sq_lo;
    } else if (lane == 16) {
      const long row = Rbase + v + 8;
      flat_idx[row] = bidx[v];
      idx_out[row]  = bidx[v];
      dists[row]    = bval[v] + sq_hi;
    }
  }

  // ---- Gather winning codebook rows into codes (float4 per lane) ----
  #pragma unroll
  for (int rr = 0; rr < 16; ++rr) {
    const int bi = __shfl(bidx[rr & 7], (rr < 8) ? 0 : 16, 32);
    const long row = Rbase + rr;
    const float4 cv = *(const float4*)(cb + (size_t)bi * C_DIM + lane * 4);
    *(float4*)(codes + row * C_DIM + lane * 4) = cv;
  }
}

// ---------------------------------------------------------------------------
extern "C" void kernel_launch(void* const* d_in, const int* in_sizes, int n_in,
                              void* d_out, int out_size, void* d_ws, size_t ws_size,
                              hipStream_t stream) {
  const float* z_e_x    = (const float*)d_in[0];  // [65536,128] fp32 (flattened)
  const float* codebook = (const float*)d_in[1];  // [1024,128]  fp32

  const int BN = in_sizes[0] / C_DIM;  // 65536 rows

  // Workspace layout
  __bf16* cbhi = (__bf16*)d_ws;                                      // 256 KB
  __bf16* cblo = (__bf16*)((char*)d_ws + (size_t)K_CB * C_DIM * 2);  // 256 KB
  float*  cbsq = (float*)((char*)d_ws + (size_t)K_CB * C_DIM * 4);   // 4 KB

  // Output layout: codes | flat_idx | idx | distances
  float* codes    = (float*)d_out;
  int*   flat_idx = (int*)d_out + (size_t)BN * C_DIM;
  int*   idx_out  = flat_idx + BN;
  float* dists    = (float*)(idx_out + BN);

  vq_prep_kernel<<<K_CB, 128, 0, stream>>>(codebook, cbhi, cblo, cbsq);

  const int rows_per_block = 8 * 16;  // 8 waves x 16 rows
  vq_search_kernel<<<BN / rows_per_block, 256, 0, stream>>>(
      z_e_x, codebook, cbhi, cblo, cbsq, codes, flat_idx, idx_out, dists);
}
